// EncDecConvLSTM_68418829025480
// MI455X (gfx1250) — matrix operand order repaired and checked by reading, not compiled, not run on hardware
//
#include <hip/hip_runtime.h>

// ---------------------------------------------------------------------------
// EncDec ConvLSTM for MI455X (gfx1250), wave32 + WMMA f16->f32.
//
// Gate conv as implicit GEMM: M = B*H*W = 16384, N = 256, K = 72*9 = 648
// (padded to 672 = 21 chunks of 32). v_wmma_f32_16x16x32_f16 per tile.
// Per-wave 2x2 tile blocking (32M x 32N), im2col patch staged in LDS.
// ---------------------------------------------------------------------------

typedef _Float16 v16h __attribute__((ext_vector_type(16)));
typedef _Float16 v8h  __attribute__((ext_vector_type(8)));
typedef float    v8f  __attribute__((ext_vector_type(8)));

#define HS      64          // spatial H = W
#define SP      4096        // HS*HS
#define BATCH   4
#define TSTEPS  10
#define FCH     8           // input feature channels
#define HD      64          // hidden dim
#define CIN     72          // FCH + HD
#define KREAL_G 648         // CIN * 9
#define KF_G    672         // padded to multiple of 32 (21 chunks)
#define NCH_G   256         // 4*HD gate channels
#define KF_O    576         // HD * 9 (already multiple of 32: 18 chunks)
#define NCH_O   16          // out channels padded 8 -> 16

__device__ __forceinline__ float sigmoidf(float x) {
    return 1.0f / (1.0f + __expf(-x));
}

// --------------------------- weight packing --------------------------------
// Wpk[n][k], k = tap*CIN + c (tap = dy*3+dx), zero-padded to KF_G.
__global__ void pack_gates_w(const float* __restrict__ W, _Float16* __restrict__ Wpk) {
    int i = blockIdx.x * blockDim.x + threadIdx.x;
    if (i >= NCH_G * KF_G) return;
    int n = i / KF_G;
    int k = i - n * KF_G;
    float v = 0.0f;
    if (k < KREAL_G) {
        int tt = k / CIN;
        int c  = k - tt * CIN;
        int dy = tt / 3, dx = tt - dy * 3;
        v = W[((n * CIN + c) * 3 + dy) * 3 + dx];
    }
    Wpk[i] = (_Float16)v;
}

__global__ void pack_out_w(const float* __restrict__ W, _Float16* __restrict__ Wpk) {
    int i = blockIdx.x * blockDim.x + threadIdx.x;
    if (i >= NCH_O * KF_O) return;
    int n = i / KF_O;
    int k = i - n * KF_O;
    int tt = k / HD;
    int c  = k - tt * HD;
    int dy = tt / 3, dx = tt - dy * 3;
    float v = (n < FCH) ? W[((n * HD + c) * 3 + dy) * 3 + dx] : 0.0f;
    Wpk[i] = (_Float16)v;
}

__global__ void zero_f32(float* __restrict__ p, int n) {
    int i = blockIdx.x * blockDim.x + threadIdx.x;
    if (i < n) p[i] = 0.0f;
}

// --------------------------- gate GEMM -------------------------------------
// grid: (512, 2). block: 128 threads = 4 waves.
// Block tile: 32 M-rows (one half image row: fixed b,y, x0..x0+31) x 128 N.
// Each wave: 32x32 (2x2 WMMA tiles). A staged in LDS as im2col [32][672] f16.
__device__ __forceinline__ void store_tile_bias(float* __restrict__ dst, const v8f& a, float bb) {
    *(float4*)dst       = make_float4(a[0] + bb, a[1] + bb, a[2] + bb, a[3] + bb);
    *(float4*)(dst + 4) = make_float4(a[4] + bb, a[5] + bb, a[6] + bb, a[7] + bb);
}

__global__ __launch_bounds__(128)
void gates_gemm(const float* __restrict__ x_seq, int t,
                const float* __restrict__ h,
                const _Float16* __restrict__ Wpk,
                const float* __restrict__ bias,
                float* __restrict__ gates)
{
    __shared__ __align__(16) _Float16 As[32][KF_G];   // 43008 B

    const int tid = threadIdx.x;
    const int mb  = blockIdx.x;            // 0..511
    const int b   = mb >> 7;               // 128 blocks per batch image
    const int y   = (mb & 127) >> 1;
    const int x0  = (mb & 1) << 5;

    // ---- im2col fill: 32 rows x 672 K (exact 168 iters per thread) ----
    for (int i = tid; i < 32 * KF_G; i += 128) {
        int m = i / KF_G;
        int k = i - m * KF_G;
        float v = 0.0f;
        if (k < KREAL_G) {
            int tt = k / CIN;
            int c  = k - tt * CIN;
            int dy = tt / 3 - 1;
            int dx = tt - (tt / 3) * 3 - 1;
            int yy = y + dy, xx = x0 + m + dx;
            if ((unsigned)yy < HS && (unsigned)xx < HS) {
                v = (c < FCH)
                    ? x_seq[(((b * TSTEPS + t) * FCH + c) << 12) + (yy << 6) + xx]
                    : h[((b * HD + (c - FCH)) << 12) + (yy << 6) + xx];
            }
        }
        As[m][k] = (_Float16)v;
    }
    __syncthreads();

    const int wave  = tid >> 5;
    const int lane  = tid & 31;
    const int lr    = lane & 15;
    const int lh    = lane >> 4;
    const int nbase = (blockIdx.y << 7) + (wave << 5);

    v8f acc00 = {}, acc01 = {}, acc10 = {}, acc11 = {};
    const _Float16* B0 = Wpk + (size_t)(nbase      + lr) * KF_G;
    const _Float16* B1 = Wpk + (size_t)(nbase + 16 + lr) * KF_G;

    for (int kc = 0; kc < KF_G / 32; ++kc) {
        const int kb = kc << 5;
        union { v16h v; v8h h2[2]; } a0, a1, b0, b1;
        // A 16x32 f16 layout: low 8 halfs = K kb+8*lh.., high 8 = K kb+16+8*lh..
        a0.h2[0] = *(const v8h*)&As[lr     ][kb +      8 * lh];
        a0.h2[1] = *(const v8h*)&As[lr     ][kb + 16 + 8 * lh];
        a1.h2[0] = *(const v8h*)&As[16 + lr][kb +      8 * lh];
        a1.h2[1] = *(const v8h*)&As[16 + lr][kb + 16 + 8 * lh];
        // B 32x16 f16 layout: lane = col, low 8 halfs = K kb+8*lh.., high 8 = K kb+16+8*lh..
        b0.h2[0] = *(const v8h*)(B0 + kb +      8 * lh);
        b0.h2[1] = *(const v8h*)(B0 + kb + 16 + 8 * lh);
        b1.h2[0] = *(const v8h*)(B1 + kb +      8 * lh);
        b1.h2[1] = *(const v8h*)(B1 + kb + 16 + 8 * lh);
        acc00 = __builtin_amdgcn_wmma_f32_16x16x32_f16(false, a0.v, false, b0.v, (short)0, acc00, false, false);
        acc01 = __builtin_amdgcn_wmma_f32_16x16x32_f16(false, a0.v, false, b1.v, (short)0, acc01, false, false);
        acc10 = __builtin_amdgcn_wmma_f32_16x16x32_f16(false, a1.v, false, b0.v, (short)0, acc10, false, false);
        acc11 = __builtin_amdgcn_wmma_f32_16x16x32_f16(false, a1.v, false, b1.v, (short)0, acc11, false, false);
    }

    // C/D layout: lane<16 -> M=r, lane>=16 -> M=8+r, N=lane%16.
    // Each lane owns 8 consecutive x positions at fixed channel -> float4 x2.
    const int ch0 = nbase + lr;
    const int ch1 = nbase + 16 + lr;
    const float bb0 = bias[ch0];
    const float bb1 = bias[ch1];
    float* base = gates + ((size_t)b * NCH_G << 12) + (y << 6) + x0 + 8 * lh;
    store_tile_bias(base + ((size_t)ch0 << 12),      acc00, bb0);
    store_tile_bias(base + ((size_t)ch1 << 12),      acc01, bb1);
    store_tile_bias(base + ((size_t)ch0 << 12) + 16, acc10, bb0);
    store_tile_bias(base + ((size_t)ch1 << 12) + 16, acc11, bb1);
}

// --------------------------- LSTM pointwise --------------------------------
__global__ void lstm_update(const float* __restrict__ gates,
                            float* __restrict__ c, float* __restrict__ h)
{
    int idx = blockIdx.x * blockDim.x + threadIdx.x;   // 0 .. B*HD*SP-1
    int b   = idx >> 18;                               // HD*SP = 262144
    int rem = idx & 262143;
    int ch  = rem >> 12;
    int sp  = rem & 4095;
    const float* g = gates + (((size_t)b * NCH_G + ch) << 12) + sp;
    float f  = g[0];
    float in = g[(size_t)64  << 12];
    float gg = g[(size_t)128 << 12];
    float o  = g[(size_t)192 << 12];
    float cn = c[idx] * sigmoidf(f) + sigmoidf(in) * tanhf(gg);
    c[idx] = cn;
    h[idx] = tanhf(cn) * sigmoidf(o);
}

// --------------------------- output conv (64 -> 8, relu) -------------------
// grid: 512, block: 64 threads = 2 waves, each wave one 16x16 tile (N padded 16).
__global__ __launch_bounds__(64)
void out_conv(const float* __restrict__ h,
              const _Float16* __restrict__ Wpk,
              const float* __restrict__ bias,
              float* __restrict__ out, int t)
{
    __shared__ __align__(16) _Float16 As[32][KF_O];   // 36864 B

    const int tid = threadIdx.x;
    const int mb  = blockIdx.x;
    const int b   = mb >> 7;
    const int y   = (mb & 127) >> 1;
    const int x0  = (mb & 1) << 5;

    for (int i = tid; i < 32 * KF_O; i += 64) {        // exact 288 iters
        int m = i / KF_O;
        int k = i - m * KF_O;
        int tt = k / HD;
        int c  = k - tt * HD;
        int dy = tt / 3 - 1;
        int dx = tt - (tt / 3) * 3 - 1;
        int yy = y + dy, xx = x0 + m + dx;
        float v = 0.0f;
        if ((unsigned)yy < HS && (unsigned)xx < HS)
            v = h[((b * HD + c) << 12) + (yy << 6) + xx];
        As[m][k] = (_Float16)v;
    }
    __syncthreads();

    const int wave = tid >> 5;
    const int lane = tid & 31;
    const int lr   = lane & 15;
    const int lh   = lane >> 4;

    v8f acc = {};
    const _Float16* B0 = Wpk + (size_t)lr * KF_O;
    for (int kc = 0; kc < KF_O / 32; ++kc) {
        const int kb = kc << 5;
        union { v16h v; v8h h2[2]; } a, bb;
        a.h2[0] = *(const v8h*)&As[wave * 16 + lr][kb +      8 * lh];
        a.h2[1] = *(const v8h*)&As[wave * 16 + lr][kb + 16 + 8 * lh];
        bb.h2[0] = *(const v8h*)(B0 + kb +      8 * lh);
        bb.h2[1] = *(const v8h*)(B0 + kb + 16 + 8 * lh);
        acc = __builtin_amdgcn_wmma_f32_16x16x32_f16(false, a.v, false, bb.v, (short)0, acc, false, false);
    }

    if (lr < FCH) {                                    // only real channels
        float bv = bias[lr];
        float* p = out + ((((size_t)b * TSTEPS + t) * FCH + lr) << 12)
                       + (y << 6) + x0 + wave * 16 + 8 * lh;
        *(float4*)p       = make_float4(fmaxf(acc[0] + bv, 0.f), fmaxf(acc[1] + bv, 0.f),
                                        fmaxf(acc[2] + bv, 0.f), fmaxf(acc[3] + bv, 0.f));
        *(float4*)(p + 4) = make_float4(fmaxf(acc[4] + bv, 0.f), fmaxf(acc[5] + bv, 0.f),
                                        fmaxf(acc[6] + bv, 0.f), fmaxf(acc[7] + bv, 0.f));
    }
}

// --------------------------- driver ----------------------------------------
extern "C" void kernel_launch(void* const* d_in, const int* in_sizes, int n_in,
                              void* d_out, int out_size, void* d_ws, size_t ws_size,
                              hipStream_t stream)
{
    const float* enc_in = (const float*)d_in[0];
    const float* dec_in = (const float*)d_in[1];
    const float* enc_W  = (const float*)d_in[2];
    const float* enc_b  = (const float*)d_in[3];
    const float* dec_W  = (const float*)d_in[4];
    const float* dec_b  = (const float*)d_in[5];
    const float* out_W  = (const float*)d_in[6];
    const float* out_b  = (const float*)d_in[7];
    float* out = (float*)d_out;

    // Workspace layout (~25 MB): h | c | gates | packed f16 weights
    float* h     = (float*)d_ws;                         // 1,048,576 f32
    float* c     = h + (size_t)BATCH * HD * SP;          // 1,048,576 f32
    float* gates = c + (size_t)BATCH * HD * SP;          // 4,194,304 f32
    _Float16* WpkE = (_Float16*)(gates + (size_t)BATCH * NCH_G * SP);
    _Float16* WpkD = WpkE + (size_t)NCH_G * KF_G;
    _Float16* WpkO = WpkD + (size_t)NCH_G * KF_G;

    const int stateN = 2 * BATCH * HD * SP;              // h and c contiguous
    zero_f32<<<(stateN + 255) / 256, 256, 0, stream>>>(h, stateN);
    pack_gates_w<<<(NCH_G * KF_G + 255) / 256, 256, 0, stream>>>(enc_W, WpkE);
    pack_gates_w<<<(NCH_G * KF_G + 255) / 256, 256, 0, stream>>>(dec_W, WpkD);
    pack_out_w<<<(NCH_O * KF_O + 255) / 256, 256, 0, stream>>>(out_W, WpkO);

    dim3 ggrid(512, 2);
    const int upN = BATCH * HD * SP;
    for (int t = 0; t < TSTEPS; ++t) {                   // encoder
        gates_gemm<<<ggrid, 128, 0, stream>>>(enc_in, t, h, WpkE, enc_b, gates);
        lstm_update<<<upN / 256, 256, 0, stream>>>(gates, c, h);
    }
    for (int t = 0; t < TSTEPS; ++t) {                   // decoder
        gates_gemm<<<ggrid, 128, 0, stream>>>(dec_in, t, h, WpkD, dec_b, gates);
        lstm_update<<<upN / 256, 256, 0, stream>>>(gates, c, h);
        out_conv<<<512, 64, 0, stream>>>(h, WpkO, out_b, out, t);
    }
}
